// MulLayer_48687749267530
// MI455X (gfx1250) — matrix operand order repaired and checked by reading, not compiled, not run on hardware
//
#include <hip/hip_runtime.h>

typedef __attribute__((ext_vector_type(16))) __bf16 v16bf;
typedef __attribute__((ext_vector_type(8)))  __bf16 v8bf;
typedef __attribute__((ext_vector_type(8)))  float  v8f;
typedef __attribute__((ext_vector_type(4)))  float  f32x4;
typedef __attribute__((ext_vector_type(4)))  unsigned int u32x4;
typedef __attribute__((ext_vector_type(4)))  int    i32x4;
typedef __attribute__((ext_vector_type(8)))  int    i32x8;

#if __has_builtin(__builtin_amdgcn_tensor_load_to_lds)
#define HAVE_TDM 1
#else
#define HAVE_TDM 0
#endif

__device__ __forceinline__ v8f wmma_bf16(v16bf a, v16bf b, v8f c) {
  return __builtin_amdgcn_wmma_f32_16x16x32_bf16(false, a, false, b, (short)0, c, false, false);
}
// A-matrix (16x32 bf16) per-lane K index for fragment element e (contiguous k = 16*(lane/16) + e, same order as B)
__device__ __forceinline__ int kA(int e, int lane) {
  return ((lane >> 4) << 4) + e;
}

#if HAVE_TDM
// 1-D contiguous DMA: global -> LDS via Tensor Data Mover (D# per ISA ch.8).
// Issue from a single wave; caller must barrier afterwards. ndwords <= 65535.
__device__ __forceinline__ void tdm_load_1d(const void* gsrc, void* ldst, int ndwords) {
  unsigned long long ga = (unsigned long long)(uintptr_t)gsrc;
  unsigned lo = (unsigned)(uintptr_t)ldst;  // low 32 bits of flat addr == LDS byte offset
  u32x4 g0;
  g0[0] = 1u;                                                   // count=1, user mode
  g0[1] = lo;                                                   // lds_addr
  g0[2] = (unsigned)ga;                                         // global_addr[31:0]
  g0[3] = ((unsigned)(ga >> 32) & 0x1FFFFFFu) | (2u << 30);     // global_addr[56:32] | type=2
  i32x8 g1;
  g1[0] = 2 << 16;                          // data_size = 4 bytes
  g1[1] = (ndwords & 0xFFFF) << 16;         // tensor_dim0[15:0]
  g1[2] = (ndwords >> 16) | (1 << 16);      // tensor_dim0[31:16] | tensor_dim1 = 1
  g1[3] = (ndwords & 0xFFFF) << 16;         // tile_dim0 = ndwords
  g1[4] = 1;                                // tile_dim1 = 1
  g1[5] = ndwords;                          // tensor_dim0_stride lo32
  g1[6] = 0;
  g1[7] = 0;
  i32x4 z4 = {0, 0, 0, 0};
#if __clang_major__ >= 23
  i32x8 z8 = {0, 0, 0, 0, 0, 0, 0, 0};
  __builtin_amdgcn_tensor_load_to_lds(g0, g1, z4, z4, z8, 0);
#else
  __builtin_amdgcn_tensor_load_to_lds(g0, g1, z4, z4, 0);
#endif
  __builtin_amdgcn_s_wait_tensorcnt((short)0);
}
#endif

#define HWPIX 65536  // 256*256 spatial positions of cF/sF

// one-time bf16 conversion of the two GEMM weight matrices (enables TDM reload per block)
__global__ void wprep_kernel(const float* __restrict__ Wc, const float* __restrict__ Wu,
                             __bf16* __restrict__ Wcb, __bf16* __restrict__ Wub) {
  int t = blockIdx.x * 256 + threadIdx.x;
  if (t < 64 * 256) Wcb[t] = (__bf16)Wc[t];
  if (t < 256 * 64) Wub[t] = (__bf16)Wu[t];
}

// ---------------------------------------------------------------------------
// X = compress_w(64x256) @ cF(256x65536) + b   (WMMA bf16, f32 accum)
// ---------------------------------------------------------------------------
__global__ void compress_gemm_kernel(const float* __restrict__ cF, const __bf16* __restrict__ Wcb,
                                     const float* __restrict__ bias, float* __restrict__ X) {
  __shared__ __attribute__((aligned(64))) __bf16 Wlds[64][256];  // 32 KB (row-major, A layout)
  __shared__ __attribute__((aligned(64))) __bf16 Bcol[128][32];  // 8 KB  (column-major, B layout)
  const int tid = threadIdx.x, lane = tid & 31, wave = tid >> 5;  // 8 waves
#if HAVE_TDM
  if (wave == 0) tdm_load_1d(Wcb, &Wlds[0][0], 64 * 256 / 2);
#else
  for (int s = tid; s < 64 * 256; s += 256) Wlds[s >> 8][s & 255] = Wcb[s];
#endif
  const int p0 = blockIdx.x * 128;
  const int col = p0 + wave * 16 + (lane & 15);
  const int sn = tid & 127, skh = tid >> 7;  // staging: column, K-half
  const v8f vz = {0.f,0.f,0.f,0.f,0.f,0.f,0.f,0.f};
  v8f acc[4];
#pragma unroll
  for (int m = 0; m < 4; ++m) acc[m] = vz;
  for (int k0 = 0; k0 < 256; k0 += 32) {
    __syncthreads();
    {  // stage cF tile 32(K)x128(N) as bf16, column-major; one 32B vector store per thread
      v16bf tmp;
#pragma unroll
      for (int j = 0; j < 16; ++j)
        tmp[j] = (__bf16)cF[(size_t)(k0 + skh * 16 + j) * HWPIX + p0 + sn];
      *(v16bf*)&Bcol[sn][skh * 16] = tmp;
    }
    __syncthreads();
    v16bf bfr = *(const v16bf*)&Bcol[wave * 16 + (lane & 15)][(lane >> 4) << 4];
#pragma unroll
    for (int m = 0; m < 4; ++m) {
      v16bf afr;
#pragma unroll
      for (int e = 0; e < 16; ++e) afr[e] = Wlds[m * 16 + (lane & 15)][k0 + kA(e, lane)];
      acc[m] = wmma_bf16(afr, bfr, acc[m]);
    }
  }
#pragma unroll
  for (int m = 0; m < 4; ++m)
#pragma unroll
    for (int r = 0; r < 8; ++r) {
      int row = m * 16 + r + ((lane >> 4) << 3);
      X[(size_t)row * HWPIX + col] = acc[m][r] + bias[row];
    }
}

__global__ void rowmean_kernel(const float* __restrict__ X, float* __restrict__ rowmean) {
  const int r = blockIdx.x;
  __shared__ float red[256];
  float s = 0.f;
  for (int p = threadIdx.x; p < HWPIX; p += 256) s += X[(size_t)r * HWPIX + p];
  red[threadIdx.x] = s; __syncthreads();
  for (int st = 128; st > 0; st >>= 1) {
    if (threadIdx.x < st) red[threadIdx.x] += red[threadIdx.x + st];
    __syncthreads();
  }
  if (threadIdx.x == 0) rowmean[r] = red[0] * (1.f / 65536.f);
}

__global__ void meansub_kernel(float* __restrict__ X, const float* __restrict__ rowmean) {
  size_t idx = (size_t)blockIdx.x * 256 + threadIdx.x;
  X[idx] -= rowmean[idx >> 16];
}

// ---------------------------------------------------------------------------
// Region counts: full-res (256x256) and cnn-res (128x128 nearest subsample)
// cnt layout: [0..8]=s_full [9..17]=c_full [18..26]=s_128 [27..35]=c_128
// ---------------------------------------------------------------------------
__global__ void counts_kernel(const int* __restrict__ smasks, const int* __restrict__ cmasks,
                              int* __restrict__ cnt) {
  const int i = blockIdx.x, set = blockIdx.y;
  const int* m = (set ? cmasks : smasks) + i * HWPIX;
  __shared__ int red[256];
  int cf = 0;
  for (int p = threadIdx.x; p < HWPIX; p += 256) cf += (m[p] == 1);
  red[threadIdx.x] = cf; __syncthreads();
  for (int st = 128; st > 0; st >>= 1) {
    if (threadIdx.x < st) red[threadIdx.x] += red[threadIdx.x + st];
    __syncthreads();
  }
  if (threadIdx.x == 0) cnt[set * 9 + i] = red[0];
  __syncthreads();
  int ch = 0;
  for (int t = threadIdx.x; t < 16384; t += 256) {
    int r = t >> 7, c = t & 127;
    ch += (m[(r * 2) * 256 + c * 2] == 1);
  }
  red[threadIdx.x] = ch; __syncthreads();
  for (int st = 128; st > 0; st >>= 1) {
    if (threadIdx.x < st) red[threadIdx.x] += red[threadIdx.x + st];
    __syncthreads();
  }
  if (threadIdx.x == 0) cnt[18 + set * 9 + i] = red[0];
}

__global__ void active_kernel(const int* __restrict__ cnt, int* __restrict__ active) {
  int i = threadIdx.x;
  if (i < 9)
    active[i] = (cnt[i] >= 10) && (cnt[9 + i] >= 10) && (cnt[18 + i] >= 10) && (cnt[27 + i] >= 10);
}

// owner[p] = largest active region covering column p (9 = none); group columns by owner
__global__ void owner_kernel(const int* __restrict__ cmasks, const int* __restrict__ active,
                             int* __restrict__ owner, int* __restrict__ ocnt, int* __restrict__ colidx) {
  int p = blockIdx.x * 256 + threadIdx.x;
  int o = 9;
#pragma unroll
  for (int i = 0; i < 9; ++i)
    if (active[i] && cmasks[i * HWPIX + p] == 1) o = i;
  owner[p] = o;
  int pos = atomicAdd(&ocnt[o], 1);
  colidx[o * HWPIX + pos] = p;
}

__global__ void smbits_kernel(const int* __restrict__ smasks, const int* __restrict__ active,
                              unsigned int* __restrict__ smbits) {
  int p = blockIdx.x * 256 + threadIdx.x;
  unsigned v = 0;
#pragma unroll
  for (int i = 0; i < 9; ++i)
    if (active[i] && smasks[i * HWPIX + p] == 1) v |= (1u << i);
  smbits[p] = v;
}

// single pass over sF (64 MB): partial per-region channel sums, no float atomics
__global__ void smean_acc_kernel(const float* __restrict__ sF, const unsigned int* __restrict__ smbits,
                                 float* __restrict__ ssump) {
  const int c = blockIdx.x;
  const int p0 = blockIdx.y * 4096;
  float loc[9];
#pragma unroll
  for (int i = 0; i < 9; ++i) loc[i] = 0.f;
  for (int p = p0 + threadIdx.x; p < p0 + 4096; p += 256) {
    float v = sF[(size_t)c * HWPIX + p];
    unsigned b = smbits[p];
#pragma unroll
    for (int i = 0; i < 9; ++i)
      if ((b >> i) & 1u) loc[i] += v;
  }
  __shared__ float red[256];
  for (int i = 0; i < 9; ++i) {
    red[threadIdx.x] = loc[i]; __syncthreads();
    for (int st = 128; st > 0; st >>= 1) {
      if (threadIdx.x < st) red[threadIdx.x] += red[threadIdx.x + st];
      __syncthreads();
    }
    if (threadIdx.x == 0) ssump[blockIdx.y * 2304 + i * 256 + c] = red[0];
    __syncthreads();
  }
}

__global__ void smean_fin_kernel(const float* __restrict__ ssump, const int* __restrict__ cnt,
                                 const int* __restrict__ active, float* __restrict__ sMean) {
  int i = blockIdx.x, c = threadIdx.x;
  float s = 0.f;
  for (int t = 0; t < 16; ++t) s += ssump[t * 2304 + i * 256 + c];
  sMean[i * 256 + c] = active[i] ? s / (float)cnt[i] : 0.f; // cnt[i] = smask full-res count
}

// ---------------------------------------------------------------------------
// 3-layer strided CNN (VALU; FLOP-trivial and memory-bound)
// ---------------------------------------------------------------------------
__global__ void conv1_kernel(const float* __restrict__ img, const float* __restrict__ w,
                             const float* __restrict__ b, __bf16* __restrict__ f1) {
  __shared__ float Wl[64 * 27];
  __shared__ float Bl[64];
  for (int s = threadIdx.x; s < 64 * 27; s += 256) Wl[s] = w[s];
  if (threadIdx.x < 64) Bl[threadIdx.x] = b[threadIdx.x];
  __syncthreads();
  int idx = blockIdx.x * 256 + threadIdx.x;          // 512*512 outputs
  int oy = idx >> 9, ox = idx & 511;
  float in[27];
#pragma unroll
  for (int ci = 0; ci < 3; ++ci)
#pragma unroll
    for (int dy = 0; dy < 3; ++dy)
#pragma unroll
      for (int dx = 0; dx < 3; ++dx) {
        int iy = oy * 2 + dy - 1, ix = ox * 2 + dx - 1;
        bool ok = (iy >= 0) & (iy < 1024) & (ix >= 0) & (ix < 1024);
        in[ci * 9 + dy * 3 + dx] = ok ? img[ci * 1048576 + iy * 1024 + ix] : 0.f;
      }
  for (int co = 0; co < 64; ++co) {
    float a = Bl[co];
#pragma unroll
    for (int j = 0; j < 27; ++j) a += Wl[co * 27 + j] * in[j];
    f1[(size_t)co * 262144 + idx] = (__bf16)fmaxf(a, 0.f);
  }
}

__global__ void conv2_kernel(const __bf16* __restrict__ f1, const float* __restrict__ w,
                             const float* __restrict__ b, float* __restrict__ f2) {
  __shared__ __bf16 Wl[32 * 576];  // 36 KB
  for (int s = threadIdx.x; s < 32 * 576; s += 256) Wl[s] = (__bf16)w[s];
  __syncthreads();
  int idx = blockIdx.x * 256 + threadIdx.x;          // 256*256 outputs
  int oy = idx >> 8, ox = idx & 255;
  float acc[32];
#pragma unroll
  for (int co = 0; co < 32; ++co) acc[co] = b[co];
  for (int ci = 0; ci < 64; ++ci)
#pragma unroll
    for (int dy = 0; dy < 3; ++dy)
#pragma unroll
      for (int dx = 0; dx < 3; ++dx) {
        int iy = oy * 2 + dy - 1, ix = ox * 2 + dx - 1;
        bool ok = (iy >= 0) & (iy < 512) & (ix >= 0) & (ix < 512);
        float v = ok ? (float)f1[(size_t)ci * 262144 + iy * 512 + ix] : 0.f;
        int wb = ci * 9 + dy * 3 + dx;
#pragma unroll
        for (int co = 0; co < 32; ++co) acc[co] += v * (float)Wl[co * 576 + wb];
      }
  for (int co = 0; co < 32; ++co) f2[(size_t)co * HWPIX + idx] = fmaxf(acc[co], 0.f);
}

__global__ void conv3_kernel(const float* __restrict__ f2, const float* __restrict__ w,
                             const float* __restrict__ b, float* __restrict__ f3) {
  __shared__ float Wl[16 * 288];   // 18 KB
  for (int s = threadIdx.x; s < 16 * 288; s += 256) Wl[s] = w[s];
  __syncthreads();
  int idx = blockIdx.x * 256 + threadIdx.x;          // 128*128 outputs
  int oy = idx >> 7, ox = idx & 127;
  float acc[16];
#pragma unroll
  for (int co = 0; co < 16; ++co) acc[co] = b[co];
  for (int ci = 0; ci < 32; ++ci)
#pragma unroll
    for (int dy = 0; dy < 3; ++dy)
#pragma unroll
      for (int dx = 0; dx < 3; ++dx) {
        int iy = oy * 2 + dy - 1, ix = ox * 2 + dx - 1;
        bool ok = (iy >= 0) & (iy < 256) & (ix >= 0) & (ix < 256);
        float v = ok ? f2[(size_t)ci * HWPIX + iy * 256 + ix] : 0.f;
        int wb = ci * 9 + dy * 3 + dx;
#pragma unroll
        for (int co = 0; co < 16; ++co) acc[co] += v * Wl[co * 288 + wb];
      }
  for (int co = 0; co < 16; ++co) f3[(size_t)co * 16384 + idx] = acc[co];
}

// ordered compaction of the 128x128 mask into rank -> position (pooling bins need order)
__global__ void compact128_kernel(const int* __restrict__ masks, int* __restrict__ idxsel) {
  const int i = blockIdx.x;
  const int* m = masks + i * HWPIX;
  __shared__ int wsum[8];
  __shared__ int running;
  if (threadIdx.x == 0) running = 0;
  __syncthreads();
  for (int base = 0; base < 16384; base += 256) {
    int t = base + threadIdx.x;
    int r = t >> 7, c = t & 127;
    int mv = (m[(r * 2) * 256 + c * 2] == 1) ? 1 : 0;
    unsigned bal32 = (unsigned)__ballot(mv);   // wave32: low 32 bits
    int lane = threadIdx.x & 31;
    int wv = threadIdx.x >> 5;
    unsigned ltmask = (lane == 0) ? 0u : (0xFFFFFFFFu >> (32 - lane));
    int pre = __popc(bal32 & ltmask);
    if (lane == 0) wsum[wv] = __popc(bal32);
    __syncthreads();
    int woff = running;
    for (int q = 0; q < wv; ++q) woff += wsum[q];
    if (mv) idxsel[i * 16384 + woff + pre] = t;
    __syncthreads();
    if (threadIdx.x == 0) {
      int tot = 0;
      for (int q = 0; q < 8; ++q) tot += wsum[q];
      running += tot;
    }
    __syncthreads();
  }
}

// adaptive max pool to 256 bins; one thread per bin, 16 channels in registers
__global__ void pool_kernel(const float* __restrict__ f3, const int* __restrict__ idxsel,
                            const int* __restrict__ cnt128, float* __restrict__ pooled) {
  const int i = blockIdx.x;
  const int j = threadIdx.x;
  const int N = cnt128[i];
  float* outp = pooled + i * 4096;
  if (N < 10) {
#pragma unroll
    for (int c = 0; c < 16; ++c) outp[c * 256 + j] = 0.f;
    return;
  }
  const int start = (j * N) >> 8;
  const int end = ((j + 1) * N + 255) >> 8;
  float acc[16];
#pragma unroll
  for (int c = 0; c < 16; ++c) acc[c] = -__builtin_inff();
  for (int t = start; t < end; ++t) {
    int p = idxsel[i * 16384 + t];
#pragma unroll
    for (int c = 0; c < 16; ++c) acc[c] = fmaxf(acc[c], f3[c * 16384 + p]);
  }
#pragma unroll
  for (int c = 0; c < 16; ++c) outp[c * 256 + j] = acc[c];
}

// ---------------------------------------------------------------------------
// mats = fcw(4096x4096) @ pooled-batch(4096x16pad) + fcb  (fcw streams once/image)
// ---------------------------------------------------------------------------
__global__ void fc_gemm_kernel(const float* __restrict__ fcw, const float* __restrict__ fcb,
                               const float* __restrict__ pooled, const int* __restrict__ cnt128,
                               float* __restrict__ mats) {
  __shared__ __attribute__((aligned(64))) __bf16 Bcol[16][32];  // column-major B tile
  const int tid = threadIdx.x, lane = tid & 31, wave = tid >> 5;  // 4 waves
  const int row0 = blockIdx.x * 64 + wave * 16;
  const int mrow = row0 + (lane & 15);
  v8f acc = {0.f,0.f,0.f,0.f,0.f,0.f,0.f,0.f};
  for (int k0 = 0; k0 < 4096; k0 += 32) {
    __syncthreads();
    if (tid < 32) {
      int n = tid & 15, kh = tid >> 4;
      bool use = (n < 9) && (cnt128[n] >= 10);
      v16bf tmp;
#pragma unroll
      for (int j = 0; j < 16; ++j)
        tmp[j] = (__bf16)(use ? pooled[n * 4096 + k0 + kh * 16 + j] : 0.f);
      *(v16bf*)&Bcol[n][kh * 16] = tmp;
    }
    __syncthreads();
    __builtin_prefetch(&fcw[(size_t)mrow * 4096 + k0 + 64], 0, 1);
    const float* arow = fcw + (size_t)mrow * 4096 + k0 + ((lane >> 4) << 4);
    const f32x4* a4 = (const f32x4*)arow;          // 32B aligned (k0 % 32 == 0)
    f32x4 q0 = a4[0], q1 = a4[1], q2 = a4[2], q3 = a4[3];
    v16bf afr;
#pragma unroll
    for (int j = 0; j < 4; ++j) {
      afr[j]      = (__bf16)q0[j];
      afr[4 + j]  = (__bf16)q1[j];
      afr[8 + j]  = (__bf16)q2[j];
      afr[12 + j] = (__bf16)q3[j];
    }
    v16bf bfr = *(const v16bf*)&Bcol[lane & 15][(lane >> 4) << 4];
    acc = wmma_bf16(afr, bfr, acc);
  }
  const int n = lane & 15;
  if (n < 9) {
#pragma unroll
    for (int r = 0; r < 8; ++r) {
      int m2 = row0 + r + ((lane >> 4) << 3);
      mats[n * 4096 + m2] = acc[r] + fcb[m2];
    }
  }
}

// T_i = sMat_i(64x64) @ cMat_i(64x64), stored bf16 for the trans GEMM
__global__ void tmat_kernel(const float* __restrict__ mats, const int* __restrict__ active,
                            __bf16* __restrict__ Tbf) {
  int i = blockIdx.x;
  if (!active[i]) return;
  const float* S = mats + i * 4096;             // style image (img 0)
  const float* C = mats + 9 * 4096 + i * 4096;  // content image (img 1)
  for (int s = threadIdx.x; s < 4096; s += 256) {
    int r = s >> 6, c2 = s & 63;
    float a = 0.f;
    for (int k = 0; k < 64; ++k) a += S[r * 64 + k] * C[k * 64 + c2];
    Tbf[i * 4096 + s] = (__bf16)a;
  }
}

// ---------------------------------------------------------------------------
// trans[:,p] = T_owner(p) @ X[:,p]  — owner-grouped gathered WMMA GEMM
// ---------------------------------------------------------------------------
__global__ void trans_kernel(const __bf16* __restrict__ Tbf, const float* __restrict__ X,
                             const int* __restrict__ ocnt, const int* __restrict__ colidx,
                             float* __restrict__ trans) {
  const int o = blockIdx.y;
  const int cntv = ocnt[o];
  const int n0 = blockIdx.x * 128;
  if (n0 >= cntv) return;
  const int tid = threadIdx.x, lane = tid & 31, wave = tid >> 5;
  if (o == 9) {  // no owner: trans column = X column
    for (int s = tid; s < 128 * 64; s += 256) {
      int nn = s & 127, r = s >> 7;
      int idx = n0 + nn;
      if (idx < cntv) {
        int p = colidx[9 * HWPIX + idx];
        trans[(size_t)r * HWPIX + p] = X[(size_t)r * HWPIX + p];
      }
    }
    return;
  }
  __shared__ __attribute__((aligned(64))) __bf16 Tlds[64][64];  // 8 KB
#if HAVE_TDM
  if (wave == 0) tdm_load_1d(Tbf + o * 4096, &Tlds[0][0], 4096 / 2);
#else
  for (int s = tid; s < 4096; s += 256) Tlds[s >> 6][s & 63] = Tbf[o * 4096 + s];
#endif
  __syncthreads();
  int nIdx = n0 + wave * 16 + (lane & 15);
  bool valid = nIdx < cntv;
  int p = colidx[o * HWPIX + (valid ? nIdx : (cntv - 1))];  // clamp keeps loads safe
  const v8f vz = {0.f,0.f,0.f,0.f,0.f,0.f,0.f,0.f};
  v8f acc[4];
#pragma unroll
  for (int m = 0; m < 4; ++m) acc[m] = vz;
#pragma unroll
  for (int ks = 0; ks < 2; ++ks) {
    v16bf bfr;
#pragma unroll
    for (int e = 0; e < 16; ++e)
      bfr[e] = (__bf16)X[(size_t)(ks * 32 + ((lane >> 4) << 4) + e) * HWPIX + p];
#pragma unroll
    for (int m = 0; m < 4; ++m) {
      v16bf afr;
#pragma unroll
      for (int e = 0; e < 16; ++e)
        afr[e] = Tlds[m * 16 + (lane & 15)][ks * 32 + kA(e, lane)];
      acc[m] = wmma_bf16(afr, bfr, acc[m]);
    }
  }
  if (valid) {
#pragma unroll
    for (int m = 0; m < 4; ++m)
#pragma unroll
      for (int r = 0; r < 8; ++r) {
        int row = m * 16 + r + ((lane >> 4) << 3);
        trans[(size_t)row * HWPIX + p] = acc[m][r];
      }
  }
}

// ---------------------------------------------------------------------------
// out = unzip_w(256x64) @ trans + unzip_b + sMean[owner]   (fused WMMA)
// ---------------------------------------------------------------------------
__global__ void unzip_kernel(const float* __restrict__ trans, const __bf16* __restrict__ Wub,
                             const float* __restrict__ bu, const float* __restrict__ sMean,
                             const int* __restrict__ owner, float* __restrict__ out) {
  __shared__ __attribute__((aligned(64))) __bf16 Wlds[256][64];  // 32 KB (row-major A)
  __shared__ __attribute__((aligned(64))) __bf16 Bcol[32][64];   // 4 KB (column-major B)
  const int tid = threadIdx.x, lane = tid & 31, wave = tid >> 5;  // 8 waves
#if HAVE_TDM
  if (wave == 0) tdm_load_1d(Wub, &Wlds[0][0], 256 * 64 / 2);
#else
  for (int s = tid; s < 256 * 64; s += 256) Wlds[s >> 6][s & 63] = Wub[s];
#endif
  const int p0 = blockIdx.x * 32;
  {  // stage trans tile 64(K)x32(N) as bf16, column-major; 16B vector store per thread
    int scol = tid & 31, skh = tid >> 5;  // 0..7
    v8bf tmp;
#pragma unroll
    for (int j = 0; j < 8; ++j)
      tmp[j] = (__bf16)trans[(size_t)(skh * 8 + j) * HWPIX + p0 + scol];
    *(v8bf*)&Bcol[scol][skh * 8] = tmp;
  }
  __syncthreads();
  const int ntile = wave & 1;
  const int mg = (wave >> 1) * 4;  // 4 M-tiles per wave
  const int col = p0 + ntile * 16 + (lane & 15);
  const v8f vz = {0.f,0.f,0.f,0.f,0.f,0.f,0.f,0.f};
  v8f acc[4];
#pragma unroll
  for (int m = 0; m < 4; ++m) acc[m] = vz;
#pragma unroll
  for (int ks = 0; ks < 2; ++ks) {
    v16bf bfr = *(const v16bf*)&Bcol[ntile * 16 + (lane & 15)][ks * 32 + ((lane >> 4) << 4)];
#pragma unroll
    for (int mi = 0; mi < 4; ++mi) {
      v16bf afr;
#pragma unroll
      for (int e = 0; e < 16; ++e)
        afr[e] = Wlds[(mg + mi) * 16 + (lane & 15)][ks * 32 + kA(e, lane)];
      acc[mi] = wmma_bf16(afr, bfr, acc[mi]);
    }
  }
  const int o = owner[col];
#pragma unroll
  for (int mi = 0; mi < 4; ++mi)
#pragma unroll
    for (int r = 0; r < 8; ++r) {
      int row = (mg + mi) * 16 + r + ((lane >> 4) << 3);
      float add = bu[row] + (o < 9 ? sMean[o * 256 + row] : 0.f);
      out[(size_t)row * HWPIX + col] = acc[mi][r] + add;
    }
}

// ---------------------------------------------------------------------------
extern "C" void kernel_launch(void* const* d_in, const int* in_sizes, int n_in,
                              void* d_out, int out_size, void* d_ws, size_t ws_size,
                              hipStream_t stream) {
  (void)in_sizes; (void)n_in; (void)out_size; (void)ws_size;
  const float* cF         = (const float*)d_in[0];
  const float* sF         = (const float*)d_in[1];
  const float* content    = (const float*)d_in[2];
  const float* style      = (const float*)d_in[3];
  const int*   cmasks     = (const int*)d_in[4];
  const int*   smasks     = (const int*)d_in[5];
  const float* compress_w = (const float*)d_in[6];
  const float* compress_b = (const float*)d_in[7];
  const float* unzip_w    = (const float*)d_in[8];
  const float* unzip_b    = (const float*)d_in[9];
  float* out = (float*)d_out;

  char* w = (char*)d_ws;
  size_t off = 0;
  auto take = [&](size_t bytes) -> char* {
    char* p = w + off;
    off += (bytes + 255) & ~(size_t)255;
    return p;
  };
  float*        X      = (float*)take(64ull * HWPIX * 4);
  float*        trans  = (float*)take(64ull * HWPIX * 4);
  float*        rowmean= (float*)take(64 * 4);
  int*          cnt    = (int*)take(36 * 4);
  int*          active = (int*)take(9 * 4);
  int*          owner  = (int*)take((size_t)HWPIX * 4);
  int*          ocnt   = (int*)take(10 * 4);
  int*          colidx = (int*)take(10ull * HWPIX * 4);
  unsigned int* smbits = (unsigned int*)take((size_t)HWPIX * 4);
  float*        ssump  = (float*)take(16ull * 9 * 256 * 4);
  float*        sMean  = (float*)take(9 * 256 * 4);
  __bf16*       f1     = (__bf16*)take(64ull * 262144 * 2);
  float*        f2     = (float*)take(32ull * HWPIX * 4);
  float*        f3     = (float*)take(16ull * 16384 * 4);
  int*          idxsel = (int*)take(9ull * 16384 * 4);
  float*        pooled = (float*)take(2ull * 9 * 4096 * 4);
  float*        mats   = (float*)take(2ull * 9 * 4096 * 4);
  __bf16*       Tbf    = (__bf16*)take(9ull * 4096 * 2);
  __bf16*       Wcb    = (__bf16*)take(64ull * 256 * 2);
  __bf16*       Wub    = (__bf16*)take(256ull * 64 * 2);

  hipMemsetAsync(ocnt, 0, 10 * 4, stream);

  // one-time bf16 weight conversion (feeds the TDM per-block LDS loads)
  wprep_kernel<<<64, 256, 0, stream>>>(compress_w, unzip_w, Wcb, Wub);

  // X = compress GEMM, then mean-subtract
  compress_gemm_kernel<<<512, 256, 0, stream>>>(cF, Wcb, compress_b, X);
  rowmean_kernel<<<64, 256, 0, stream>>>(X, rowmean);
  meansub_kernel<<<16384, 256, 0, stream>>>(X, rowmean);

  // region gating, column ownership, style means (one pass over sF)
  counts_kernel<<<dim3(9, 2), 256, 0, stream>>>(smasks, cmasks, cnt);
  active_kernel<<<1, 32, 0, stream>>>(cnt, active);
  owner_kernel<<<256, 256, 0, stream>>>(cmasks, active, owner, ocnt, colidx);
  smbits_kernel<<<256, 256, 0, stream>>>(smasks, active, smbits);
  smean_acc_kernel<<<dim3(256, 16), 256, 0, stream>>>(sF, smbits, ssump);
  smean_fin_kernel<<<9, 256, 0, stream>>>(ssump, cnt, active, sMean);

  // CNN + pooling + batched FC per image (style=0, content=1), conv buffers reused
  for (int img = 0; img < 2; ++img) {
    const float* image = img ? content : style;
    const int*   masks = img ? cmasks : smasks;
    const float* w1  = (const float*)d_in[10 + img * 8 + 0];
    const float* b1  = (const float*)d_in[10 + img * 8 + 1];
    const float* w2  = (const float*)d_in[10 + img * 8 + 2];
    const float* b2  = (const float*)d_in[10 + img * 8 + 3];
    const float* w3  = (const float*)d_in[10 + img * 8 + 4];
    const float* b3  = (const float*)d_in[10 + img * 8 + 5];
    const float* fcw = (const float*)d_in[10 + img * 8 + 6];
    const float* fcb = (const float*)d_in[10 + img * 8 + 7];
    conv1_kernel<<<1024, 256, 0, stream>>>(image, w1, b1, f1);
    conv2_kernel<<<256, 256, 0, stream>>>(f1, w2, b2, f2);
    conv3_kernel<<<64, 256, 0, stream>>>(f2, w3, b3, f3);
    compact128_kernel<<<9, 256, 0, stream>>>(masks, idxsel);
    const int* cnt128 = cnt + 18 + img * 9;
    pool_kernel<<<9, 256, 0, stream>>>(f3, idxsel, cnt128, pooled + img * 9 * 4096);
    fc_gemm_kernel<<<64, 128, 0, stream>>>(fcw, fcb, pooled + img * 9 * 4096, cnt128,
                                           mats + img * 9 * 4096);
  }

  // T = sM @ cM; trans = T_owner @ X (owner-grouped); out = unzip + bias + sMean
  tmat_kernel<<<9, 256, 0, stream>>>(mats, active, Tbf);
  trans_kernel<<<dim3(512, 10), 256, 0, stream>>>(Tbf, X, ocnt, colidx, trans);
  unzip_kernel<<<2048, 256, 0, stream>>>(trans, unzip_w ? Wub : Wub, unzip_b, sMean, owner, out);
}
